// GAT_23235773071553
// MI455X (gfx1250) — matrix operand order repaired and checked by reading, not compiled, not run on hardware
//
#include <hip/hip_runtime.h>
#include <hip/hip_bf16.h>
#include <math.h>

#define N_BATCH 4096
#define T_SEQ   60
#define F_IN    64
#define H_DIM   128
#define G3      384
#define D_FF    256

typedef __attribute__((ext_vector_type(16))) __bf16 v16bf;
typedef __attribute__((ext_vector_type(8)))  float  v8f;

static __device__ __forceinline__ v8f wmma_bf16(v16bf a, v16bf b, v8f c) {
    // D = A(16x32) * B(32x16) + C, f32 accumulate
    return __builtin_amdgcn_wmma_f32_16x16x32_bf16(
        /*neg_a=*/false, a, /*neg_b=*/false, b,
        /*c_mod=*/(short)0, c, /*reuse_a=*/false, /*reuse_b=*/false);
}

// Fast activations on the TRANS pipe (v_exp_f32 + v_rcp_f32 co-execute with
// XDL WMMA); ~1 ulp, far below bf16 GEMM noise. Avoids IEEE div fixup chain.
static __device__ __forceinline__ float fast_rcp(float x) {
    return __builtin_amdgcn_rcpf(x);
}
static __device__ __forceinline__ float sigmoidf_(float x) {
    return fast_rcp(1.0f + __expf(-x));
}
static __device__ __forceinline__ float tanhf_(float x) {
    // 2*sigmoid(2x)-1 ; correct saturation at +/-inf through rcp
    return 2.0f * fast_rcp(1.0f + __expf(-2.0f * x)) - 1.0f;
}

// ---- A-fragment builders (16-bit A 16x32 layout, ISA 7.12.2) -------------
// lane<16: row=lane,   K = {kb..kb+7, kb+16..kb+23} with kb = kt*32
// lane>=16: row=lane-16, same but shifted by 8 (half = lane>>4)
static __device__ __forceinline__ v16bf a_frag_from_lds(const float* h16x128,
                                                        int lane, int kt) {
    int row = lane & 15;
    int s1  = kt * 32 + ((lane >> 4) << 3);
    const float* p = h16x128 + row * H_DIM;
    v16bf f;
#pragma unroll
    for (int i = 0; i < 8; ++i) f[i] = (__bf16)p[s1 + i];
#pragma unroll
    for (int i = 0; i < 8; ++i) f[8 + i] = (__bf16)p[s1 + 16 + i];
    return f;
}

static __device__ __forceinline__ v16bf a_frag_from_x(const float* __restrict__ x,
                                                      int rowBase, int t,
                                                      int lane, int kt) {
    int row = lane & 15;
    const float* p = x + ((size_t)(rowBase + row) * T_SEQ + t) * F_IN;
    int s1 = kt * 32 + ((lane >> 4) << 3);
    v16bf f;
#pragma unroll
    for (int i = 0; i < 8; ++i) f[i] = (__bf16)p[s1 + i];
#pragma unroll
    for (int i = 0; i < 8; ++i) f[8 + i] = (__bf16)p[s1 + 16 + i];
    return f;
}

// ---- fragment-ready packed B weights ------------------------------------
// frag index = ((nt*KT + kt)*32 + lane)*16, contiguous 32B per lane
static __device__ __forceinline__ v16bf b_frag(const __bf16* __restrict__ pw,
                                               int nt, int kt, int KT, int lane) {
    return *(const v16bf*)(pw + ((((size_t)nt * KT + kt) * 32 + lane) << 4));
}

// Pack W (G3 x K, row-major, f32) -> bf16 B-fragments.
// B[k][n] = W[n][k];  half h = lane>>4: i<8 -> k = kt*32 + 8h + i, i>=8 -> k = kt*32 + 16 + 8h + (i-8)
__global__ void pack_weights_kernel(const float* __restrict__ W,
                                    __bf16* __restrict__ dst, int KT) {
    int total = 24 * KT * 512;
    int idx = blockIdx.x * blockDim.x + threadIdx.x;
    if (idx >= total) return;
    int i    = idx & 15;
    int lane = (idx >> 4) & 31;
    int kt   = (idx >> 9) % KT;
    int nt   = idx / (KT * 512);
    int K    = KT * 32;
    int n = nt * 16 + (lane & 15);
    int k = kt * 32 + ((lane >> 4) << 3) + (i & 7) + ((i >> 3) << 4);
    dst[idx] = (__bf16)W[(size_t)n * K + k];
}

// ---- one GRU layer step for a 16-row wave tile --------------------------
template <int KTI>
static __device__ __forceinline__ void gru_layer_step(
    float* hlds,                       // this wave's h: [16][128] in LDS
    const v16bf* Ai,                   // input A-fragments (KTI of them)
    const __bf16* __restrict__ pWi,    // packed W_ih (K = KTI*32)
    const v16bf* Ah,                   // h A-fragments (4 of them)
    const __bf16* __restrict__ pWh,    // packed W_hh (K = 128)
    const float* __restrict__ bi, const float* __restrict__ bh, int lane) {
    const int colL  = lane & 15;
    const int rbase = (lane >> 4) << 3;
    const v8f zero = {0.f, 0.f, 0.f, 0.f, 0.f, 0.f, 0.f, 0.f};
    for (int j = 0; j < 8; ++j) {
        v8f gir = zero, giz = zero, gin = zero;
        v8f ghr = zero, ghz = zero, ghn = zero;
#pragma unroll
        for (int kt = 0; kt < KTI; ++kt) {
            gir = wmma_bf16(Ai[kt], b_frag(pWi, j,      kt, KTI, lane), gir);
            giz = wmma_bf16(Ai[kt], b_frag(pWi, j + 8,  kt, KTI, lane), giz);
            gin = wmma_bf16(Ai[kt], b_frag(pWi, j + 16, kt, KTI, lane), gin);
        }
#pragma unroll
        for (int kt = 0; kt < 4; ++kt) {
            ghr = wmma_bf16(Ah[kt], b_frag(pWh, j,      kt, 4, lane), ghr);
            ghz = wmma_bf16(Ah[kt], b_frag(pWh, j + 8,  kt, 4, lane), ghz);
            ghn = wmma_bf16(Ah[kt], b_frag(pWh, j + 16, kt, 4, lane), ghn);
        }
        int col = j * 16 + colL;
        float bir = bi[col], biz = bi[H_DIM + col], bin = bi[2 * H_DIM + col];
        float bhr = bh[col], bhz = bh[H_DIM + col], bhn = bh[2 * H_DIM + col];
#pragma unroll
        for (int v = 0; v < 8; ++v) {
            float r = sigmoidf_(gir[v] + bir + ghr[v] + bhr);
            float z = sigmoidf_(giz[v] + biz + ghz[v] + bhz);
            float n = tanhf_(gin[v] + bin + r * (ghn[v] + bhn));
            float* hp = hlds + (rbase + v) * H_DIM + col;
            float hold = *hp;
            *hp = (1.0f - z) * n + z * hold;   // safe: A-frags already in regs
        }
    }
}

// ---- persistent 2-layer GRU: each wave owns 16 batch rows ---------------
__global__ __launch_bounds__(128) void gru_kernel(
    const float* __restrict__ x,
    const float* __restrict__ bi0, const float* __restrict__ bh0,
    const float* __restrict__ bi1, const float* __restrict__ bh1,
    const __bf16* __restrict__ pWi0, const __bf16* __restrict__ pWh0,
    const __bf16* __restrict__ pWi1, const __bf16* __restrict__ pWh1,
    float* __restrict__ hT_out) {
    __shared__ float sh[2][4][16 * H_DIM];   // 64 KB: h0/h1 per wave
    const int lane = threadIdx.x & 31;
    const int wave = threadIdx.x >> 5;
    const int rowBase = blockIdx.x * 64 + wave * 16;
    float* h0 = &sh[0][wave][0];
    float* h1 = &sh[1][wave][0];
    for (int i = lane; i < 16 * H_DIM; i += 32) { h0[i] = 0.f; h1[i] = 0.f; }
    // wave-private LDS: DS ops are in-order per wave -> no barriers needed
    for (int t = 0; t < T_SEQ; ++t) {
        v16bf Ax[2];
#pragma unroll
        for (int kt = 0; kt < 2; ++kt) Ax[kt] = a_frag_from_x(x, rowBase, t, lane, kt);
        v16bf Ah0[4];
#pragma unroll
        for (int kt = 0; kt < 4; ++kt) Ah0[kt] = a_frag_from_lds(h0, lane, kt);
        gru_layer_step<2>(h0, Ax, pWi0, Ah0, pWh0, bi0, bh0, lane);

        v16bf Ax1[4];
#pragma unroll
        for (int kt = 0; kt < 4; ++kt) Ax1[kt] = a_frag_from_lds(h0, lane, kt); // updated h0
        v16bf Ah1[4];
#pragma unroll
        for (int kt = 0; kt < 4; ++kt) Ah1[kt] = a_frag_from_lds(h1, lane, kt);
        gru_layer_step<4>(h1, Ax1, pWi1, Ah1, pWh1, bi1, bh1, lane);
    }
    for (int i = lane; i < 16 * H_DIM; i += 32) {
        int r = i >> 7, c = i & (H_DIM - 1);
        hT_out[(size_t)(rowBase + r) * H_DIM + c] = h1[i];
    }
}

// ---- xt = hT@Wt^T + bt; s_j = xt@a[:H]; s_i = xt@a[H:] ------------------
__global__ __launch_bounds__(128) void attn_scores_kernel(
    const float* __restrict__ hT, const float* __restrict__ Wt,
    const float* __restrict__ bt, const float* __restrict__ a,
    float* __restrict__ s_i, float* __restrict__ s_j) {
    int nrow = blockIdx.x, c = threadIdx.x;
    const float* hp = hT + (size_t)nrow * H_DIM;
    const float* wp = Wt + (size_t)c * H_DIM;
    float acc = bt[c];
    for (int k = 0; k < H_DIM; ++k) acc += hp[k] * wp[k];
    __shared__ float pj[H_DIM], pi[H_DIM];
    pj[c] = acc * a[c];
    pi[c] = acc * a[H_DIM + c];
    __syncthreads();
    for (int s = 64; s > 0; s >>= 1) {
        if (c < s) { pj[c] += pj[c + s]; pi[c] += pi[c + s]; }
        __syncthreads();
    }
    if (c == 0) { s_j[nrow] = pj[0]; s_i[nrow] = pi[0]; }
}

__global__ void smax_kernel(const float* __restrict__ s_j, float* __restrict__ smax) {
    __shared__ float m[256];
    float v = -INFINITY;
    for (int i = threadIdx.x; i < N_BATCH; i += 256) v = fmaxf(v, s_j[i]);
    m[threadIdx.x] = v;
    __syncthreads();
    for (int s = 128; s > 0; s >>= 1) {
        if (threadIdx.x < s) m[threadIdx.x] = fmaxf(m[threadIdx.x], m[threadIdx.x + s]);
        __syncthreads();
    }
    if (threadIdx.x == 0) *smax = m[0];
}

// ---- hidden2 = softmax(lrelu(s_i+s_j)) @ hT + hT ------------------------
// row max m_i = lrelu(s_i + max_j s_j) (lrelu monotone) -> single pass
__global__ __launch_bounds__(256) void attn_apply_kernel(
    const float* __restrict__ hT, const float* __restrict__ s_i,
    const float* __restrict__ s_j, const float* __restrict__ smax_p,
    float* __restrict__ hidden2) {
    __shared__ float hch[64][H_DIM];   // 32 KB chunk of hT rows
    __shared__ float sjc[64];
    __shared__ float wch[8][8][64];    // per-wave softmax weights
    const int lane = threadIdx.x & 31;
    const int wave = threadIdx.x >> 5;
    const int rowBase = blockIdx.x * 64 + wave * 8;
    const float smax = *smax_p;
    float si[8], mi[8], di[8], acc[8][4];
#pragma unroll
    for (int r = 0; r < 8; ++r) {
        si[r] = s_i[rowBase + r];
        float m = si[r] + smax;
        mi[r] = m > 0.f ? m : 0.01f * m;
        di[r] = 0.f;
        acc[r][0] = acc[r][1] = acc[r][2] = acc[r][3] = 0.f;
    }
    for (int j0 = 0; j0 < N_BATCH; j0 += 64) {
        __syncthreads();
        for (int i = threadIdx.x; i < 64 * H_DIM / 4; i += 256)
            ((float4*)&hch[0][0])[i] = ((const float4*)(hT + (size_t)j0 * H_DIM))[i];
        if (threadIdx.x < 64) sjc[threadIdx.x] = s_j[j0 + threadIdx.x];
        __syncthreads();
        // phase A: each lane computes exp once per (row, j)
#pragma unroll
        for (int r = 0; r < 8; ++r) {
#pragma unroll
            for (int h = 0; h < 2; ++h) {
                int jj = lane + h * 32;
                float sc = si[r] + sjc[jj];
                sc = sc > 0.f ? sc : 0.01f * sc;
                wch[wave][r][jj] = __expf(sc - mi[r]);
            }
        }
        // phase B: weighted accumulate (wch is wave-private, DS in-order)
        for (int jj = 0; jj < 64; ++jj) {
            float hv0 = hch[jj][lane];
            float hv1 = hch[jj][lane + 32];
            float hv2 = hch[jj][lane + 64];
            float hv3 = hch[jj][lane + 96];
#pragma unroll
            for (int r = 0; r < 8; ++r) {
                float w = wch[wave][r][jj];
                di[r] += w;
                acc[r][0] += w * hv0; acc[r][1] += w * hv1;
                acc[r][2] += w * hv2; acc[r][3] += w * hv3;
            }
        }
    }
#pragma unroll
    for (int r = 0; r < 8; ++r) {
        size_t n = rowBase + r;
        float inv = fast_rcp(di[r]);
        const float* hres = hT + n * H_DIM;
        float* op = hidden2 + n * H_DIM;
        op[lane]      = acc[r][0] * inv + hres[lane];
        op[lane + 32] = acc[r][1] * inv + hres[lane + 32];
        op[lane + 64] = acc[r][2] * inv + hres[lane + 64];
        op[lane + 96] = acc[r][3] * inv + hres[lane + 96];
    }
}

__global__ __launch_bounds__(128) void fc_lrelu_kernel(
    const float* __restrict__ hidden2, const float* __restrict__ Wfc,
    const float* __restrict__ bfc, float* __restrict__ hidden3) {
    int nrow = blockIdx.x, c = threadIdx.x;
    const float* hp = hidden2 + (size_t)nrow * H_DIM;
    const float* wp = Wfc + (size_t)c * H_DIM;
    float acc = bfc[c];
    for (int k = 0; k < H_DIM; ++k) acc += hp[k] * wp[k];
    acc = acc > 0.f ? acc : 0.01f * acc;
    hidden3[(size_t)nrow * H_DIM + c] = acc;
}

__global__ __launch_bounds__(256) void mlp1_gelu_kernel(
    const float* __restrict__ hidden3, const float* __restrict__ Wp1,
    const float* __restrict__ bp1, float* __restrict__ h1g) {
    int nrow = blockIdx.x, d = threadIdx.x;
    const float* hp = hidden3 + (size_t)nrow * H_DIM;
    const float* wp = Wp1 + (size_t)d * H_DIM;
    float acc = bp1[d];
    for (int k = 0; k < H_DIM; ++k) acc += hp[k] * wp[k];
    // exact gelu: 0.5*x*(1+erf(x/sqrt(2)))
    float g = 0.5f * acc * (1.0f + erff(acc * 0.70710678118654752f));
    h1g[(size_t)nrow * D_FF + d] = g;
}

__global__ __launch_bounds__(256) void out_kernel(
    const float* __restrict__ h1g, const float* __restrict__ Wp2,
    const float* __restrict__ bp2, float* __restrict__ out) {
    int lane = threadIdx.x & 31;
    int wave = threadIdx.x >> 5;
    int n = blockIdx.x * 8 + wave;
    const float* hp = h1g + (size_t)n * D_FF;
    float p = 0.f;
    for (int d = lane; d < D_FF; d += 32) p += hp[d] * Wp2[d];
#pragma unroll
    for (int s = 16; s > 0; s >>= 1) p += __shfl_xor(p, s, 32);
    if (lane == 0) out[n] = p + bp2[0];
}

extern "C" void kernel_launch(void* const* d_in, const int* in_sizes, int n_in,
                              void* d_out, int out_size, void* d_ws, size_t ws_size,
                              hipStream_t stream) {
    (void)in_sizes; (void)n_in; (void)out_size; (void)ws_size;
    const float* x    = (const float*)d_in[0];
    const float* Wih0 = (const float*)d_in[1];
    const float* Whh0 = (const float*)d_in[2];
    const float* bih0 = (const float*)d_in[3];
    const float* bhh0 = (const float*)d_in[4];
    const float* Wih1 = (const float*)d_in[5];
    const float* Whh1 = (const float*)d_in[6];
    const float* bih1 = (const float*)d_in[7];
    const float* bhh1 = (const float*)d_in[8];
    const float* Wt   = (const float*)d_in[9];
    const float* bt   = (const float*)d_in[10];
    const float* a    = (const float*)d_in[11];
    const float* Wfc  = (const float*)d_in[12];
    const float* bfc  = (const float*)d_in[13];
    const float* Wp1  = (const float*)d_in[14];
    const float* bp1  = (const float*)d_in[15];
    const float* Wp2  = (const float*)d_in[16];
    const float* bp2  = (const float*)d_in[17];
    float* out = (float*)d_out;

    char* ws = (char*)d_ws;
    size_t off = 0;
    auto take = [&](size_t bytes) {
        char* p = ws + off;
        off = (off + bytes + 255) & ~(size_t)255;
        return p;
    };
    __bf16* pWi0 = (__bf16*)take((size_t)24 * 2 * 512 * sizeof(__bf16));
    __bf16* pWh0 = (__bf16*)take((size_t)24 * 4 * 512 * sizeof(__bf16));
    __bf16* pWi1 = (__bf16*)take((size_t)24 * 4 * 512 * sizeof(__bf16));
    __bf16* pWh1 = (__bf16*)take((size_t)24 * 4 * 512 * sizeof(__bf16));
    float* hT      = (float*)take((size_t)N_BATCH * H_DIM * 4);
    float* s_i     = (float*)take((size_t)N_BATCH * 4);
    float* s_j     = (float*)take((size_t)N_BATCH * 4);
    float* smax    = (float*)take(256);
    float* hidden2 = (float*)take((size_t)N_BATCH * H_DIM * 4);
    float* hidden3 = (float*)take((size_t)N_BATCH * H_DIM * 4);
    float* h1g     = (float*)take((size_t)N_BATCH * D_FF * 4);

    pack_weights_kernel<<<(24 * 2 * 512 + 255) / 256, 256, 0, stream>>>(Wih0, pWi0, 2);
    pack_weights_kernel<<<(24 * 4 * 512 + 255) / 256, 256, 0, stream>>>(Whh0, pWh0, 4);
    pack_weights_kernel<<<(24 * 4 * 512 + 255) / 256, 256, 0, stream>>>(Wih1, pWi1, 4);
    pack_weights_kernel<<<(24 * 4 * 512 + 255) / 256, 256, 0, stream>>>(Whh1, pWh1, 4);

    gru_kernel<<<N_BATCH / 64, 128, 0, stream>>>(x, bih0, bhh0, bih1, bhh1,
                                                 pWi0, pWh0, pWi1, pWh1, hT);

    attn_scores_kernel<<<N_BATCH, 128, 0, stream>>>(hT, Wt, bt, a, s_i, s_j);
    smax_kernel<<<1, 256, 0, stream>>>(s_j, smax);
    attn_apply_kernel<<<N_BATCH / 64, 256, 0, stream>>>(hT, s_i, s_j, smax, hidden2);
    fc_lrelu_kernel<<<N_BATCH, 128, 0, stream>>>(hidden2, Wfc, bfc, hidden3);
    mlp1_gelu_kernel<<<N_BATCH, 256, 0, stream>>>(hidden3, Wp1, bp1, h1g);
    out_kernel<<<N_BATCH / 8, 256, 0, stream>>>(h1g, Wp2, bp2, out);
}
